// token_representation_41686952575123
// MI455X (gfx1250) — matrix operand order repaired and checked by reading, not compiled, not run on hardware
//
#include <hip/hip_runtime.h>
#include <hip/hip_bf16.h>

typedef __attribute__((ext_vector_type(16))) _Float16 v16h;
typedef __attribute__((ext_vector_type(8)))  _Float16 v8h;
typedef __attribute__((ext_vector_type(8)))  float    v8f;

#define NWORDS   8192
#define CHARLEN  16
#define WORDDIM  300
#define CHARDIM  64
#define HID      128
#define GATES    512      // 4*HID
#define OUTDIM   512
#define INFODIM  556
#define KPAD     576      // 556 padded to multiple of 32
#define MTILE    32       // words per 256-thread block

// LDS row pitches in halves (pad to break 256B-stride bank conflicts)
#define WHH_PITCH 136
#define WIH_PITCH 72
#define H_PITCH   136

// ---- WMMA fragment loaders (ISA 7.12.2 layouts, wave32) -------------------
// A (16x32 f16): lane%16 = M row, elem e -> K = kbase + 8*hi + (e<8 ? e : e+8).
__device__ __forceinline__ v16h load_a_frag(const _Float16* row, int kbase, int hi) {
    const _Float16* p = row + kbase + hi * 8;
    v8h lo = *(const v8h*)(p);
    v8h hh = *(const v8h*)(p + 16);
    v16h a;
#pragma unroll
    for (int i = 0; i < 8; ++i) { a[i] = lo[i]; a[i + 8] = hh[i]; }
    return a;
}
// B (32x16 f16): lane%16 = N column, elem e -> K = kbase + 8*hi + (e<8 ? e : e+8).
// wrow points at W[col][0] (row-major W[n][k], i.e. B = W^T).
__device__ __forceinline__ v16h load_b_frag(const _Float16* wrow, int kbase, int hi) {
    const _Float16* p = wrow + kbase + hi * 8;
    v8h lo = *(const v8h*)(p);
    v8h hh = *(const v8h*)(p + 16);
    v16h b;
#pragma unroll
    for (int i = 0; i < 8; ++i) { b[i] = lo[i]; b[i + 8] = hh[i]; }
    return b;
}

// branch-free, saturating activations built on v_exp_f32
__device__ __forceinline__ float sigmoidf_(float x) { return 1.0f / (1.0f + __expf(-x)); }
__device__ __forceinline__ float tanhf_(float x)    { return 2.0f / (1.0f + __expf(-2.0f * x)) - 1.0f; }

#define WMMA_F16(a, b, c) \
    __builtin_amdgcn_wmma_f32_16x16x32_f16(false, (a), false, (b), (short)0, (c), false, false)

// ---- prep kernels ---------------------------------------------------------
__global__ void gather_x_kernel(const int* __restrict__ char_ids,
                                const float* __restrict__ char_table,
                                _Float16* __restrict__ Xf16) {
    int idx = blockIdx.x * blockDim.x + threadIdx.x;     // over N*16*64
    if (idx >= NWORDS * CHARLEN * CHARDIM) return;
    int d  = idx & (CHARDIM - 1);
    int nt = idx >> 6;                                   // n*16 + t
    int cid = char_ids[nt];
    Xf16[idx] = (_Float16)char_table[cid * CHARDIM + d];
}

__global__ void feat_word_kernel(const int* __restrict__ word_ids,
                                 const float* __restrict__ word_table,
                                 _Float16* __restrict__ feat) {
    int n = blockIdx.x;            // 8192 blocks
    int k = threadIdx.x;           // blockDim = 320
    if (k < WORDDIM) {
        feat[(size_t)n * KPAD + k] =
            (_Float16)word_table[(size_t)word_ids[n] * WORDDIM + k];
    } else {
        int kk = INFODIM + (k - WORDDIM);                // 556..575 zero pad
        feat[(size_t)n * KPAD + kk] = (_Float16)0.0f;
    }
}

__global__ void conv_wout_kernel(const float* __restrict__ Wout,
                                 _Float16* __restrict__ WoutF16) {
    int idx = blockIdx.x * blockDim.x + threadIdx.x;     // over 512*576
    if (idx >= OUTDIM * KPAD) return;
    int j = idx / KPAD, k = idx % KPAD;
    WoutF16[idx] = (k < INFODIM) ? (_Float16)Wout[j * INFODIM + k] : (_Float16)0.0f;
}

__global__ void conv_f16_kernel(const float* __restrict__ src,
                                _Float16* __restrict__ dst, int n) {
    int i = blockIdx.x * blockDim.x + threadIdx.x;
    if (i < n) dst[i] = (_Float16)src[i];
}

__global__ void bias_comb_kernel(const float* __restrict__ a,
                                 const float* __restrict__ b,
                                 float* __restrict__ dst) {
    int i = blockIdx.x * blockDim.x + threadIdx.x;
    if (i < GATES) dst[i] = a[i] + b[i];
}

// ---- fused bidirectional char-LSTM (blockIdx.y = direction) ---------------
// 256 threads = 8 waves. Wave w: column-quarter wq = w&3 (j in [32wq,32wq+32)
// across ALL four gates -> cell update is pure per-lane register math),
// word-subtile mt = w>>2 (words n0 + 16*mt .. +15).
__global__ void __launch_bounds__(256)
lstm_kernel(const _Float16* __restrict__ Xf16,
            const _Float16* __restrict__ Wih16f, const _Float16* __restrict__ Whh16f,
            const float* __restrict__ biasf,
            const _Float16* __restrict__ Wih16b, const _Float16* __restrict__ Whh16b,
            const float* __restrict__ biasb,
            _Float16* __restrict__ feat) {
    __shared__ _Float16 shWih[GATES * WIH_PITCH];   // 73.7 KB
    __shared__ _Float16 shWhh[GATES * WHH_PITCH];   // 139.3 KB
    __shared__ _Float16 shH[MTILE * H_PITCH];       // 8.7 KB

    const int tid  = threadIdx.x;
    const int lane = tid & 31;
    const int wave = tid >> 5;
    const int hi   = lane >> 4;
    const int lm   = lane & 15;
    const int wq   = wave & 3;
    const int mt   = wave >> 2;
    const int dir  = blockIdx.y;
    const int n0   = blockIdx.x * MTILE + mt * 16;   // this wave's word tile

    const _Float16* Wih16 = dir ? Wih16b : Wih16f;
    const _Float16* Whh16 = dir ? Whh16b : Whh16f;
    const float*    bias  = dir ? biasb  : biasf;

    // ---- async-DMA f16 weights into LDS (per-lane 16B chunks, ASYNCcnt) ----
    {
        unsigned whhBase = (unsigned)(size_t)(&shWhh[0]);
        for (int idx = tid; idx < GATES * (HID / 8); idx += 256) {   // 8192 chunks
            int j = idx >> 4, c = idx & 15;
            unsigned l = whhBase + (unsigned)(j * (WHH_PITCH * 2) + c * 16);
            const _Float16* g = Whh16 + idx * 8;
            asm volatile("global_load_async_to_lds_b128 %0, %1, off"
                         :: "v"(l), "v"(g) : "memory");
        }
        unsigned wihBase = (unsigned)(size_t)(&shWih[0]);
        for (int idx = tid; idx < GATES * (CHARDIM / 8); idx += 256) { // 4096 chunks
            int j = idx >> 3, c = idx & 7;
            unsigned l = wihBase + (unsigned)(j * (WIH_PITCH * 2) + c * 16);
            const _Float16* g = Wih16 + idx * 8;
            asm volatile("global_load_async_to_lds_b128 %0, %1, off"
                         :: "v"(l), "v"(g) : "memory");
        }
    }
    for (int idx = tid; idx < MTILE * H_PITCH; idx += 256) shH[idx] = (_Float16)0.0f;
    asm volatile("s_wait_asynccnt 0" ::: "memory");
    __syncthreads();

    // per-lane gate biases (4 gates x 2 column groups)
    const int colb0 = wq * 32;                       // j base for this wave
    float bI[2], bF[2], bG[2], bO[2];
#pragma unroll
    for (int g2 = 0; g2 < 2; ++g2) {
        int cb = colb0 + g2 * 16 + lm;
        bI[g2] = bias[cb];
        bF[g2] = bias[128 + cb];
        bG[g2] = bias[256 + cb];
        bO[g2] = bias[384 + cb];
    }

    float cst[16], hsum[16];
#pragma unroll
    for (int i = 0; i < 16; ++i) { cst[i] = 0.0f; hsum[i] = 0.0f; }

    // start pointer + signed stride instead of per-step t select
    const _Float16* xrow  = Xf16 + (size_t)(n0 + lm) * (CHARLEN * CHARDIM)
                          + (dir ? (CHARLEN - 1) * CHARDIM : 0);
    const int xstride     = dir ? -CHARDIM : CHARDIM;
    const _Float16* hrow  = &shH[(mt * 16 + lm) * H_PITCH];

#pragma unroll 1
    for (int step = 0; step < CHARLEN; ++step) {
        v8f acc[2][4] = {};                          // C starts at inline 0

        // input part, K=64: batch 4 B-frag loads per gate group, then 4 WMMAs
#pragma unroll
        for (int kt = 0; kt < 2; ++kt) {
            v16h a = load_a_frag(xrow, kt * 32, hi);
#pragma unroll
            for (int g2 = 0; g2 < 2; ++g2) {
                int cb = colb0 + g2 * 16 + lm;
                v16h bf[4];
#pragma unroll
                for (int q = 0; q < 4; ++q)
                    bf[q] = load_b_frag(&shWih[(q * 128 + cb) * WIH_PITCH], kt * 32, hi);
#pragma unroll
                for (int q = 0; q < 4; ++q)
                    acc[g2][q] = WMMA_F16(a, bf[q], acc[g2][q]);
            }
        }
        // recurrent part, K=128
#pragma unroll
        for (int kt = 0; kt < 4; ++kt) {
            v16h a = load_a_frag(hrow, kt * 32, hi);
#pragma unroll
            for (int g2 = 0; g2 < 2; ++g2) {
                int cb = colb0 + g2 * 16 + lm;
                v16h bf[4];
#pragma unroll
                for (int q = 0; q < 4; ++q)
                    bf[q] = load_b_frag(&shWhh[(q * 128 + cb) * WHH_PITCH], kt * 32, hi);
#pragma unroll
                for (int q = 0; q < 4; ++q)
                    acc[g2][q] = WMMA_F16(a, bf[q], acc[g2][q]);
            }
        }
        __syncthreads();                             // all shH reads done

        // cell update: all four gate pre-activations live in this lane
#pragma unroll
        for (int g2 = 0; g2 < 2; ++g2) {
#pragma unroll
            for (int r = 0; r < 8; ++r) {
                float pi = acc[g2][0][r] + bI[g2];
                float pf = acc[g2][1][r] + bF[g2];
                float pg = acc[g2][2][r] + bG[g2];
                float po = acc[g2][3][r] + bO[g2];
                int ci = g2 * 8 + r;
                float cc = sigmoidf_(pf) * cst[ci] + sigmoidf_(pi) * tanhf_(pg);
                cst[ci] = cc;
                float h = sigmoidf_(po) * tanhf_(cc);
                hsum[ci] += h;
                // restage h (f16) as next step's A fragment: row m, col j
                shH[(mt * 16 + r + hi * 8) * H_PITCH + colb0 + g2 * 16 + lm] = (_Float16)h;
            }
        }
        __syncthreads();                             // shH ready for next step

        xrow += xstride;
    }

    // emit summed hidden states into padded feature rows
    const int fbase = WORDDIM + dir * HID;
#pragma unroll
    for (int g2 = 0; g2 < 2; ++g2) {
#pragma unroll
        for (int r = 0; r < 8; ++r) {
            int m = r + hi * 8;
            feat[(size_t)(n0 + m) * KPAD + fbase + colb0 + g2 * 16 + lm] =
                (_Float16)hsum[g2 * 8 + r];
        }
    }
}

// ---- output projection: out = tanh(feat @ Wout^T + b) ---------------------
__global__ void __launch_bounds__(256)
proj_kernel(const _Float16* __restrict__ feat, const _Float16* __restrict__ Wout,
            const float* __restrict__ bout, float* __restrict__ out) {
    const int tid  = threadIdx.x;
    const int lane = tid & 31;
    const int wave = tid >> 5;
    const int hi   = lane >> 4;
    const int lm   = lane & 15;
    const int wq   = wave & 3;
    const int mt   = wave >> 2;
    const int n0   = blockIdx.x * MTILE + mt * 16;

    const _Float16* arow = feat + (size_t)(n0 + lm) * KPAD;

#pragma unroll 1
    for (int nt8 = 0; nt8 < 8; ++nt8) {
        const int nt  = wq * 8 + nt8;
        const int col = nt * 16 + lm;
        v8f acc = {};
        const _Float16* brow = Wout + (size_t)col * KPAD;
#pragma unroll 2
        for (int kt = 0; kt < KPAD / 32; ++kt) {     // 18 K-tiles
            v16h a = load_a_frag(arow, kt * 32, hi);
            v16h b = load_b_frag(brow, kt * 32, hi);
            acc = WMMA_F16(a, b, acc);
        }
        const float bb = bout[col];
#pragma unroll
        for (int r = 0; r < 8; ++r) {
            int m = r + hi * 8;
            out[(size_t)(n0 + m) * OUTDIM + col] = tanhf_(acc[r] + bb);
        }
    }
}

// ---- host launch ----------------------------------------------------------
extern "C" void kernel_launch(void* const* d_in, const int* in_sizes, int n_in,
                              void* d_out, int out_size, void* d_ws, size_t ws_size,
                              hipStream_t stream) {
    const int*   word_ids   = (const int*)d_in[0];
    const int*   char_ids   = (const int*)d_in[1];
    const float* word_table = (const float*)d_in[2];
    const float* char_table = (const float*)d_in[3];
    const float* Wih_f = (const float*)d_in[4];
    const float* Whh_f = (const float*)d_in[5];
    const float* bih_f = (const float*)d_in[6];
    const float* bhh_f = (const float*)d_in[7];
    const float* Wih_b = (const float*)d_in[8];
    const float* Whh_b = (const float*)d_in[9];
    const float* bih_b = (const float*)d_in[10];
    const float* bhh_b = (const float*)d_in[11];
    const float* W_out = (const float*)d_in[12];
    const float* b_out = (const float*)d_in[13];
    float* out = (float*)d_out;

    // workspace carve (all sizes 128B-aligned)
    char* w = (char*)d_ws;
    _Float16* Xf16    = (_Float16*)w; w += (size_t)NWORDS * CHARLEN * CHARDIM * 2; // 16 MB
    _Float16* featF16 = (_Float16*)w; w += (size_t)NWORDS * KPAD * 2;              // 9 MB
    _Float16* WoutF16 = (_Float16*)w; w += (size_t)OUTDIM * KPAD * 2;              // 0.56 MB
    _Float16* Wih16f  = (_Float16*)w; w += (size_t)GATES * CHARDIM * 2;
    _Float16* Whh16f  = (_Float16*)w; w += (size_t)GATES * HID * 2;
    _Float16* Wih16b  = (_Float16*)w; w += (size_t)GATES * CHARDIM * 2;
    _Float16* Whh16b  = (_Float16*)w; w += (size_t)GATES * HID * 2;
    float*    biasCf  = (float*)w;    w += (size_t)GATES * 4;
    float*    biasCb  = (float*)w;

    // prep
    {
        int total = NWORDS * CHARLEN * CHARDIM;
        gather_x_kernel<<<(total + 255) / 256, 256, 0, stream>>>(char_ids, char_table, Xf16);
    }
    feat_word_kernel<<<NWORDS, 320, 0, stream>>>(word_ids, word_table, featF16);
    {
        int total = OUTDIM * KPAD;
        conv_wout_kernel<<<(total + 255) / 256, 256, 0, stream>>>(W_out, WoutF16);
    }
    conv_f16_kernel<<<(GATES * CHARDIM + 255) / 256, 256, 0, stream>>>(Wih_f, Wih16f, GATES * CHARDIM);
    conv_f16_kernel<<<(GATES * HID     + 255) / 256, 256, 0, stream>>>(Whh_f, Whh16f, GATES * HID);
    conv_f16_kernel<<<(GATES * CHARDIM + 255) / 256, 256, 0, stream>>>(Wih_b, Wih16b, GATES * CHARDIM);
    conv_f16_kernel<<<(GATES * HID     + 255) / 256, 256, 0, stream>>>(Whh_b, Whh16b, GATES * HID);
    bias_comb_kernel<<<2, 256, 0, stream>>>(bih_f, bhh_f, biasCf);
    bias_comb_kernel<<<2, 256, 0, stream>>>(bih_b, bhh_b, biasCb);

    // bidirectional char LSTM, both directions in one grid
    {
        dim3 grid(NWORDS / MTILE, 2, 1);
        lstm_kernel<<<grid, 256, 0, stream>>>(Xf16, Wih16f, Whh16f, biasCf,
                                              Wih16b, Whh16b, biasCb, featF16);
    }

    // projection
    proj_kernel<<<NWORDS / MTILE, 256, 0, stream>>>(featF16, WoutF16, b_out, out);

    (void)in_sizes; (void)n_in; (void)out_size; (void)ws_size;
}
